// HGCNModel_64287070486573
// MI455X (gfx1250) — compile-verified, matrix operand order checked by reading, not run on hardware
//
#include <hip/hip_runtime.h>

#define EPS 1e-7f

typedef __attribute__((ext_vector_type(16))) _Float16 v16h;
typedef __attribute__((ext_vector_type(8)))  float    v8f;

// ---------------------------------------------------------------------------
// wave32 helpers
// ---------------------------------------------------------------------------
__device__ inline float wave_sum32(float x) {
#pragma unroll
  for (int m = 16; m > 0; m >>= 1) x += __shfl_xor(x, m, 32);
  return x;
}

// ---------------------------------------------------------------------------
// t = logmap0(expmap0(features))   (faithful chain, one wave per node,
// lane l owns elements l and l+32 of the 64-wide row)
// ---------------------------------------------------------------------------
__global__ __launch_bounds__(256) void k_tangent_in(const float* __restrict__ f,
                                                    float* __restrict__ t, int Nn) {
  const int lane = threadIdx.x & 31;
  const int node = blockIdx.x * 8 + (threadIdx.x >> 5);
  if (node >= Nn) return;
  const float* v = f + (long)node * 64;
  float v0 = v[lane], v1 = v[lane + 32];

  float s = wave_sum32(v0 * v0 + v1 * v1);
  float nrm = sqrtf(fmaxf(s, EPS));
  float sc  = sinhf(nrm) / nrm;
  float x0  = fmaxf(coshf(nrm), 1.0f + EPS);
  float xs0 = sc * v0, xs1 = sc * v1;
  float s2 = wave_sum32(xs0 * xs0 + xs1 * xs1);
  float nrm2 = sqrtf(fmaxf(s2, EPS));
  float sc2 = acoshf(x0) / nrm2;

  t[(long)node * 64 + lane]      = sc2 * xs0;
  t[(long)node * 64 + lane + 32] = sc2 * xs1;
}

// ---------------------------------------------------------------------------
// t_next = logmap0(expmap0(sigmoid(logmap0(expmap0(agg / max(deg,1))))))
// (covers hgcn_layer tail + sigmoid_nonlin + next layer's logmap0)
// ---------------------------------------------------------------------------
__global__ __launch_bounds__(256) void k_post_layer(const float* __restrict__ agg,
                                                    const float* __restrict__ deg,
                                                    float* __restrict__ t, int Nn) {
  const int lane = threadIdx.x & 31;
  const int node = blockIdx.x * 8 + (threadIdx.x >> 5);
  if (node >= Nn) return;
  const float d = fmaxf(deg[node], 1.0f);
  float a0 = agg[(long)node * 64 + lane] / d;
  float a1 = agg[(long)node * 64 + lane + 32] / d;

  // expmap0 -> logmap0
  float s = wave_sum32(a0 * a0 + a1 * a1);
  float nrm = sqrtf(fmaxf(s, EPS));
  float sc  = sinhf(nrm) / nrm;
  float x0  = fmaxf(coshf(nrm), 1.0f + EPS);
  float xs0 = sc * a0, xs1 = sc * a1;
  float s2 = wave_sum32(xs0 * xs0 + xs1 * xs1);
  float sc2 = acoshf(x0) / sqrtf(fmaxf(s2, EPS));
  float u0 = sc2 * xs0, u1 = sc2 * xs1;

  // sigmoid
  float g0 = 1.0f / (1.0f + expf(-u0));
  float g1 = 1.0f / (1.0f + expf(-u1));

  // expmap0 -> logmap0
  float s3 = wave_sum32(g0 * g0 + g1 * g1);
  float nrm3 = sqrtf(fmaxf(s3, EPS));
  float sc3  = sinhf(nrm3) / nrm3;
  float y0   = fmaxf(coshf(nrm3), 1.0f + EPS);
  float ys0 = sc3 * g0, ys1 = sc3 * g1;
  float s4 = wave_sum32(ys0 * ys0 + ys1 * ys1);
  float sc4 = acoshf(y0) / sqrtf(fmaxf(s4, EPS));

  t[(long)node * 64 + lane]      = sc4 * ys0;
  t[(long)node * 64 + lane + 32] = sc4 * ys1;
}

// ---------------------------------------------------------------------------
// zero a float region
// ---------------------------------------------------------------------------
__global__ __launch_bounds__(256) void k_zero(float* __restrict__ p, long n) {
  long i = (long)blockIdx.x * blockDim.x + threadIdx.x;
  if (i < n) p[i] = 0.0f;
}

// ---------------------------------------------------------------------------
// Edge scatter: agg[dst] += h[src] (64 floats, wave per edge, 2 atomics/lane),
// optionally deg[dst] += 1.  h table is L2-resident (25.6 MB < 192 MB).
// ---------------------------------------------------------------------------
__global__ __launch_bounds__(256) void k_edge_scatter(const long long* __restrict__ ei,
                                                      long E,
                                                      const float* __restrict__ h,
                                                      float* __restrict__ agg,
                                                      float* __restrict__ deg,
                                                      int addDeg) {
  const int lane = threadIdx.x & 31;
  const long e = (long)blockIdx.x * 8 + (threadIdx.x >> 5);
  if (e >= E) return;
  const long src = ei[e];
  const long dst = ei[E + e];
  const float* hs = h + src * 64;
  float* ad = agg + dst * 64;
  atomicAdd(ad + lane,      hs[lane]);
  atomicAdd(ad + lane + 32, hs[lane + 32]);
  if (addDeg && lane == 0) atomicAdd(deg + dst, 1.0f);
}

// ---------------------------------------------------------------------------
// WMMA GEMM: out[M,NC] = A[M,K] @ W[K,NC] + bias  (optionally ReLU)
// f32 in/out, f16 multiply, f32 accumulate via v_wmma_f32_16x16x32_f16.
// One wave per 16-row tile; loops n-tiles of 16; K in chunks of 32.
// A fragment layout (ISA 16-bit A 16x32): lanes 0-15 -> K 0-7 & 16-23,
// lanes 16-31 -> K 8-15 & 24-31; B mirrors with N=lane&15.
// C layout: lane (0-15,n) holds rows v (VGPR v), lanes 16-31 rows v+8.
// Column index is clamped (not masked) for the padded NC=40 tile: padded
// columns compute garbage but are never stored (WMMA columns independent).
// Row guard hoisted to one wave-uniform branch (full tile vs tail tile).
// ---------------------------------------------------------------------------
template <int K, int NC, bool RELU>
__global__ __launch_bounds__(256) void k_gemm_wmma(const float* __restrict__ A,
                                                   const float* __restrict__ W,
                                                   const float* __restrict__ bias,
                                                   float* __restrict__ out, int M) {
  constexpr int KC = K / 32;
  constexpr int NT = (NC + 15) / 16;
  const int lane = threadIdx.x & 31;
  const long tile = (long)blockIdx.x * 8 + (threadIdx.x >> 5);
  const long row0 = tile * 16;
  if (row0 >= M) return;
  const int lr = lane & 15;
  const int khalf = (lane >> 4) * 8;          // K offset for A/B; row offset for C
  const bool fullTile = (row0 + 16 <= (long)M);  // wave-uniform

  // ---- load A fragments for all K chunks (kept in VGPRs, reused per n-tile)
  long arow = row0 + lr;
  if (arow >= M) arow = M - 1;  // tail clamp
  const float2* Ap2 = (const float2*)(A + arow * (long)K);
  v16h afrag[KC];
#pragma unroll
  for (int c = 0; c < KC; ++c) {
#pragma unroll
    for (int v = 0; v < 8; ++v) {
      const int kl = ((v >> 2) * 16) + ((v & 3) * 2) + khalf;   // even
      const int k = c * 32 + kl;
      const float2 p = Ap2[k >> 1];
      afrag[c][2 * v]     = (_Float16)p.x;
      afrag[c][2 * v + 1] = (_Float16)p.y;
    }
  }

#pragma unroll
  for (int nt = 0; nt < NT; ++nt) {
    const int n = nt * 16 + lr;
    const bool nok = (NC % 16 == 0) || (n < NC);
    const int ne = nok ? n : (NC - 1);        // clamped: loads unconditional
    const float bv = bias[ne];
    v8f acc;
#pragma unroll
    for (int v = 0; v < 8; ++v) acc[v] = bv;

#pragma unroll
    for (int c = 0; c < KC; ++c) {
      v16h bfrag;
#pragma unroll
      for (int v = 0; v < 8; ++v) {
        const int kl = ((v >> 2) * 16) + ((v & 3) * 2) + khalf;
        const int k = c * 32 + kl;
        bfrag[2 * v]     = (_Float16)W[(long)k * NC + ne];
        bfrag[2 * v + 1] = (_Float16)W[(long)(k + 1) * NC + ne];
      }
      acc = __builtin_amdgcn_wmma_f32_16x16x32_f16(
          /*neg_a=*/false, afrag[c], /*neg_b=*/false, bfrag,
          /*c_mod=*/(short)0, acc, /*reuse_a=*/false, /*reuse_b=*/false);
    }

    float r[8];
#pragma unroll
    for (int v = 0; v < 8; ++v)
      r[v] = RELU ? fmaxf(acc[v], 0.0f) : acc[v];

    float* op = out + (row0 + khalf) * (long)NC + n;  // row (row0+khalf), col n
    if (fullTile) {
      if (nok) {
#pragma unroll
        for (int v = 0; v < 8; ++v) op[(long)v * NC] = r[v];
      }
    } else {
      if (nok) {
#pragma unroll
        for (int v = 0; v < 8; ++v)
          if (row0 + khalf + v < (long)M) op[(long)v * NC] = r[v];
      }
    }
  }
}

// ---------------------------------------------------------------------------
// launch
// ---------------------------------------------------------------------------
extern "C" void kernel_launch(void* const* d_in, const int* in_sizes, int n_in,
                              void* d_out, int out_size, void* d_ws, size_t ws_size,
                              hipStream_t stream) {
  const float*     features = (const float*)d_in[0];
  const long long* ei       = (const long long*)d_in[1];  // int64 per reference
  const float* W1 = (const float*)d_in[2]; const float* b1 = (const float*)d_in[3];
  const float* W2 = (const float*)d_in[4]; const float* b2 = (const float*)d_in[5];
  const float* W3 = (const float*)d_in[6]; const float* b3 = (const float*)d_in[7];
  const float* W4 = (const float*)d_in[8]; const float* b4 = (const float*)d_in[9];

  const int  Nn = in_sizes[0] / 64;      // 100000 nodes
  const long E  = (long)in_sizes[1] / 2; // 1,600,000 edges

  // workspace layout (floats): t[N*64] | h[N*128] | agg[N*64] | deg[N]
  float* tbuf = (float*)d_ws;
  float* hbuf = tbuf + (size_t)Nn * 64;
  float* aggbuf = hbuf + (size_t)Nn * 128;
  float* degbuf = aggbuf + (size_t)Nn * 64;   // contiguous after agg

  const dim3 blk(256);
  const int  nodeBlocks = (Nn + 7) / 8;
  const long edgeBlocks = (E + 7) / 8;
  const int  gemmBlocks = (((Nn + 15) / 16) + 7) / 8;

  // layer 1
  k_tangent_in<<<nodeBlocks, blk, 0, stream>>>(features, tbuf, Nn);
  k_gemm_wmma<64, 64, false><<<gemmBlocks, blk, 0, stream>>>(tbuf, W1, b1, hbuf, Nn);
  {
    long zn = (long)Nn * 65;  // agg + deg in one pass
    k_zero<<<(unsigned)((zn + 255) / 256), blk, 0, stream>>>(aggbuf, zn);
  }
  k_edge_scatter<<<(unsigned)edgeBlocks, blk, 0, stream>>>(ei, E, hbuf, aggbuf, degbuf, 1);
  k_post_layer<<<nodeBlocks, blk, 0, stream>>>(aggbuf, degbuf, tbuf, Nn);

  // layer 2 (deg unchanged; only agg re-zeroed)
  k_gemm_wmma<64, 64, false><<<gemmBlocks, blk, 0, stream>>>(tbuf, W2, b2, hbuf, Nn);
  {
    long zn = (long)Nn * 64;
    k_zero<<<(unsigned)((zn + 255) / 256), blk, 0, stream>>>(aggbuf, zn);
  }
  k_edge_scatter<<<(unsigned)edgeBlocks, blk, 0, stream>>>(ei, E, hbuf, aggbuf, degbuf, 0);
  k_post_layer<<<nodeBlocks, blk, 0, stream>>>(aggbuf, degbuf, tbuf, Nn);

  // MLP head: relu(t @ W3 + b3) @ W4 + b4
  k_gemm_wmma<64, 128, true><<<gemmBlocks, blk, 0, stream>>>(tbuf, W3, b3, hbuf, Nn);
  k_gemm_wmma<128, 40, false><<<gemmBlocks, blk, 0, stream>>>(hbuf, W4, b4, (float*)d_out, Nn);
}
